// GCN_30313879175766
// MI455X (gfx1250) — matrix operand order repaired and checked by reading, not compiled, not run on hardware
//
#include <hip/hip_runtime.h>
#include <hip/hip_bf16.h>

typedef __attribute__((ext_vector_type(16))) __bf16 v16bf;
typedef __attribute__((ext_vector_type(8)))  __bf16 v8bf;
typedef __attribute__((ext_vector_type(8)))  float  v8f;

#define N_NODES 50000
#define N_EDGES 640000
#define N_SUP   3
#define D       128
#define TOT_E   (N_SUP * N_EDGES)   // 1,920,000

// ---------------------------------------------------------------------------
// Phase 0a: split x (f32) into bf16 hi/lo pair (hi = rn(x), lo = rn(x - hi))
// ---------------------------------------------------------------------------
__global__ void k_cvt_x(const float* __restrict__ x,
                        __bf16* __restrict__ xhi, __bf16* __restrict__ xlo) {
  size_t i = (size_t)blockIdx.x * blockDim.x + threadIdx.x;   // 6,400,000 exact
  float v = x[i];
  __bf16 h = (__bf16)v;
  xhi[i] = h;
  xlo[i] = (__bf16)(v - (float)h);
}

// ---------------------------------------------------------------------------
// Phase 0b: weights [s][k][n] -> transposed bf16 hi/lo [s][n][k]
// ---------------------------------------------------------------------------
__global__ void k_cvt_w(const float* __restrict__ w,
                        __bf16* __restrict__ whi, __bf16* __restrict__ wlo) {
  int i = blockIdx.x * blockDim.x + threadIdx.x;              // 49,152 exact
  int s = i >> 14; int rem = i & 16383;
  int k = rem >> 7; int n = rem & 127;
  float v = w[i];
  __bf16 h = (__bf16)v;
  int o = (s << 14) + (n << 7) + k;
  whi[o] = h;
  wlo[o] = (__bf16)(v - (float)h);
}

// ---------------------------------------------------------------------------
// Phase 1: pre[s] = x @ W[s] via bf16 WMMA with hi/lo error compensation.
// One wave computes a 16x64 output tile (4 N-subtiles) so each A fragment
// pair is reused 4x: A-side L2 traffic drops from ~614MB to ~154MB.
// 3 supports * 3125 M-tiles * 2 N-supertiles = 18750 waves = 3125 blocks x 6.
// ---------------------------------------------------------------------------
__global__ void k_gemm(const __bf16* __restrict__ xhi, const __bf16* __restrict__ xlo,
                       const __bf16* __restrict__ whi, const __bf16* __restrict__ wlo,
                       float* __restrict__ pre) {
  const int MT  = N_NODES / 16;  // 3125
  const int NST = D / 64;        // 2 N-supertiles
  int wave = threadIdx.x >> 5;
  int lane = threadIdx.x & 31;
  int tile = blockIdx.x * 6 + wave;          // 0..18749, exact grid
  int s   = tile / (MT * NST);
  int t   = tile % (MT * NST);
  int mt  = t / NST, nst = t % NST;
  int hlf = lane >> 4;
  int l15 = lane & 15;

  const __bf16* arh = xhi + (size_t)(mt * 16 + l15) * D;   // A row (hi)
  const __bf16* arl = xlo + (size_t)(mt * 16 + l15) * D;   // A row (lo)
  // B columns ([s][n][k] layout): 4 subtiles, column n = nst*64 + j*16 + l15
  const __bf16* bbase_h = whi + ((size_t)s << 14) + (size_t)(nst * 64 + l15) * D;
  const __bf16* bbase_l = wlo + ((size_t)s << 14) + (size_t)(nst * 64 + l15) * D;

  v8f c0 = {}, c1 = {}, c2 = {}, c3 = {};
#pragma unroll
  for (int kc = 0; kc < D; kc += 32) {
    // A 16x32 bf16 layout (ISA 7.12.2): half 0 -> K {0..7,16..23}, half 1 -> K {8..15,24..31}
    v8bf a0h = *(const v8bf*)(arh + kc + hlf * 8);
    v8bf a1h = *(const v8bf*)(arh + kc + 16 + hlf * 8);
    v8bf a0l = *(const v8bf*)(arl + kc + hlf * 8);
    v8bf a1l = *(const v8bf*)(arl + kc + 16 + hlf * 8);
    v16bf ah = __builtin_shufflevector(a0h, a1h, 0,1,2,3,4,5,6,7,8,9,10,11,12,13,14,15);
    v16bf al = __builtin_shufflevector(a0l, a1l, 0,1,2,3,4,5,6,7,8,9,10,11,12,13,14,15);

#pragma unroll
    for (int j = 0; j < 4; ++j) {
      // B 32x16 bf16: half 0 -> K {0..7,16..23}, half 1 -> K {8..15,24..31}; N = lane&15
      const __bf16* bh_p = bbase_h + (size_t)j * 16 * D + kc + hlf * 8;
      const __bf16* bl_p = bbase_l + (size_t)j * 16 * D + kc + hlf * 8;
      v8bf b0h = *(const v8bf*)(bh_p);
      v8bf b1h = *(const v8bf*)(bh_p + 16);
      v8bf b0l = *(const v8bf*)(bl_p);
      v8bf b1l = *(const v8bf*)(bl_p + 16);
      v16bf bh = __builtin_shufflevector(b0h, b1h, 0,1,2,3,4,5,6,7,8,9,10,11,12,13,14,15);
      v16bf bl = __builtin_shufflevector(b0l, b1l, 0,1,2,3,4,5,6,7,8,9,10,11,12,13,14,15);

      v8f c = (j == 0) ? c0 : (j == 1) ? c1 : (j == 2) ? c2 : c3;
      // pre += xhi*whi + xlo*whi + xhi*wlo   (f32 accumulate)
      c = __builtin_amdgcn_wmma_f32_16x16x32_bf16(false, ah, false, bh, (short)0, c, false, false);
      c = __builtin_amdgcn_wmma_f32_16x16x32_bf16(false, al, false, bh, (short)0, c, false, false);
      c = __builtin_amdgcn_wmma_f32_16x16x32_bf16(false, ah, false, bl, (short)0, c, false, false);
      if (j == 0) c0 = c; else if (j == 1) c1 = c; else if (j == 2) c2 = c; else c3 = c;
    }
  }

  // C/D layout: VGPR r -> M = hlf*8 + r, N = lane&15 (+ j*16 column offset)
  float* op = pre + (size_t)s * N_NODES * D + (size_t)(mt * 16 + hlf * 8) * D + nst * 64 + l15;
#pragma unroll
  for (int r = 0; r < 8; ++r) {
    op[(size_t)r * D +  0] = c0[r];
    op[(size_t)r * D + 16] = c1[r];
    op[(size_t)r * D + 32] = c2[r];
    op[(size_t)r * D + 48] = c3[r];
  }
}

// ---------------------------------------------------------------------------
// Phase 2: CSR build over destination rows (all 3 supports merged)
// ---------------------------------------------------------------------------
__global__ void k_zero(int* __restrict__ p, int n) {
  int i = blockIdx.x * blockDim.x + threadIdx.x;
  if (i < n) p[i] = 0;
}

__global__ void k_count(const int* __restrict__ rows, int* __restrict__ counts) {
  int e = blockIdx.x * blockDim.x + threadIdx.x;   // TOT_E exact
  atomicAdd(&counts[rows[e]], 1);
}

__global__ void k_scan(const int* __restrict__ counts, int* __restrict__ offs, int n) {
  __shared__ int sh[1024];
  __shared__ int carry;
  if (threadIdx.x == 0) carry = 0;
  __syncthreads();
  for (int base = 0; base < n; base += 1024) {
    int i = base + (int)threadIdx.x;
    int v = (i < n) ? counts[i] : 0;
    sh[threadIdx.x] = v;
    __syncthreads();
    for (int off = 1; off < 1024; off <<= 1) {
      int t = ((int)threadIdx.x >= off) ? sh[threadIdx.x - off] : 0;
      __syncthreads();
      sh[threadIdx.x] += t;
      __syncthreads();
    }
    if (i < n) offs[i] = carry + sh[threadIdx.x] - v;   // exclusive
    __syncthreads();
    if (threadIdx.x == 0) carry += sh[1023];
    __syncthreads();
  }
  if (threadIdx.x == 0) offs[n] = carry;                // total = TOT_E
}

__global__ void k_copy(const int* __restrict__ a, int* __restrict__ b, int n) {
  int i = blockIdx.x * blockDim.x + threadIdx.x;
  if (i < n) b[i] = a[i];
}

__global__ void k_fill(const int* __restrict__ rows, const int* __restrict__ cols,
                       const float* __restrict__ vals, int* __restrict__ cursor,
                       int* __restrict__ pcol, float* __restrict__ pval) {
  int e = blockIdx.x * blockDim.x + threadIdx.x;   // TOT_E exact
  int s = e / N_EDGES;
  int row = rows[e];
  int pos = atomicAdd(&cursor[row], 1);
  pcol[pos] = s * N_NODES + cols[e];               // row index into pre[s*N + col]
  pval[pos] = vals[e];
}

// ---------------------------------------------------------------------------
// Phase 3: per-row gather SpMM + bias + ReLU. One wave per row (wave32:
// 4 f32/lane covers D=128). No FP atomics; pre is L2-resident (76.8MB < 192MB).
// Prefetch next gathered row (global_prefetch_b8) to hide L2 latency.
// ---------------------------------------------------------------------------
__global__ void k_spmm(const int* __restrict__ offs, const int* __restrict__ pcol,
                       const float* __restrict__ pval, const float* __restrict__ pre,
                       const float* __restrict__ bias, float* __restrict__ out) {
  int wave = threadIdx.x >> 5;
  int lane = threadIdx.x & 31;
  int row  = __builtin_amdgcn_readfirstlane(blockIdx.x * 8 + wave);  // 50000 exact
  int beg = offs[row], end = offs[row + 1];
  int l4 = lane * 4;
  float ax = 0.f, ay = 0.f, az = 0.f, aw = 0.f;
  for (int p = beg; p < end; ++p) {
    int   pc = pcol[p];
    float v  = pval[p];
    int   pn = pcol[p + (p + 1 < end ? 1 : 0)];                      // clamped lookahead
    __builtin_prefetch(pre + (size_t)pn * D + l4, 0, 0);             // global_prefetch_b8
    const float4 m = *(const float4*)(pre + (size_t)pc * D + l4);
    ax += v * m.x; ay += v * m.y; az += v * m.z; aw += v * m.w;
  }
  const float4 b = *(const float4*)(bias + l4);
  float4 o;
  o.x = fmaxf(ax + b.x, 0.f);
  o.y = fmaxf(ay + b.y, 0.f);
  o.z = fmaxf(az + b.z, 0.f);
  o.w = fmaxf(aw + b.w, 0.f);
  *(float4*)(out + (size_t)row * D + l4) = o;
}

// ---------------------------------------------------------------------------
extern "C" void kernel_launch(void* const* d_in, const int* in_sizes, int n_in,
                              void* d_out, int out_size, void* d_ws, size_t ws_size,
                              hipStream_t stream) {
  (void)in_sizes; (void)n_in; (void)out_size; (void)ws_size;

  const float* x     = (const float*)d_in[0];
  const int*   erow  = (const int*)  d_in[1];
  const int*   ecol  = (const int*)  d_in[2];
  const float* evals = (const float*)d_in[3];
  const float* wgt   = (const float*)d_in[4];
  const float* bias  = (const float*)d_in[5];
  float* out = (float*)d_out;

  // workspace carve-out (256B aligned)
  char* wp = (char*)d_ws;
  auto carve = [&](size_t bytes) -> char* {
    char* p = wp;
    wp += (bytes + 255) & ~(size_t)255;
    return p;
  };
  __bf16* xhi   = (__bf16*)carve((size_t)N_NODES * D * 2);         // 12.8 MB
  __bf16* xlo   = (__bf16*)carve((size_t)N_NODES * D * 2);         // 12.8 MB
  __bf16* whi   = (__bf16*)carve((size_t)N_SUP * D * D * 2);       // 96 KB
  __bf16* wlo   = (__bf16*)carve((size_t)N_SUP * D * D * 2);       // 96 KB
  float*  pre   = (float*) carve((size_t)N_SUP * N_NODES * D * 4); // 76.8 MB
  int*    cnts  = (int*)   carve((size_t)N_NODES * 4);
  int*    offs  = (int*)   carve((size_t)(N_NODES + 1) * 4);
  int*    curs  = (int*)   carve((size_t)N_NODES * 4);
  int*    pcol  = (int*)   carve((size_t)TOT_E * 4);               // 7.68 MB
  float*  pval  = (float*) carve((size_t)TOT_E * 4);               // 7.68 MB

  // Phase 0: precision-split conversions
  k_cvt_x<<<(N_NODES * D) / 256, 256, 0, stream>>>(x, xhi, xlo);
  k_cvt_w<<<(N_SUP * D * D) / 256, 256, 0, stream>>>(wgt, whi, wlo);

  // Phase 1: WMMA GEMM -> pre  (18750 16x64 tiles, 6 waves/block, exact)
  k_gemm<<<3125, 192, 0, stream>>>(xhi, xlo, whi, wlo, pre);

  // Phase 2: CSR build
  k_zero<<<(N_NODES + 255) / 256, 256, 0, stream>>>(cnts, N_NODES);
  k_count<<<TOT_E / 256, 256, 0, stream>>>(erow, cnts);
  k_scan<<<1, 1024, 0, stream>>>(cnts, offs, N_NODES);
  k_copy<<<(N_NODES + 255) / 256, 256, 0, stream>>>(offs, curs, N_NODES);
  k_fill<<<TOT_E / 256, 256, 0, stream>>>(erow, ecol, evals, curs, pcol, pval);

  // Phase 3: gather SpMM + bias + ReLU (one wave per row)
  k_spmm<<<N_NODES / 8, 256, 0, stream>>>(offs, pcol, pval, pre, bias, out);
}
